// OctreeConvFastBnRelu_63651415327313
// MI455X (gfx1250) — matrix operand order repaired and checked by reading, not compiled, not run on hardware
//
#include <hip/hip_runtime.h>
#include <hip/hip_bf16.h>
#include <stdint.h>

typedef __attribute__((ext_vector_type(16))) __bf16 v16bf;
typedef __attribute__((ext_vector_type(8)))  float  v8f;
typedef __attribute__((ext_vector_type(8)))  unsigned int v8u;

#define CIN   64
#define NK3   27
#define COUT  128
#define KTOT  (CIN * NK3)      // 1728
#define NTILE 128
#define KC    32               // K per WMMA chunk
#define NCHUNK (KTOT / KC)     // 54 ; chunk kc covers K = kc*32 .. kc*32+31
#define LDSROW 17              // dwords per 32-bf16 row, +1 pad (bank friendly)

// ---------------- prolog: x (C,H) f32 -> xT (H,C) bf16 ----------------
__global__ __launch_bounds__(256)
void prep_x(const float* __restrict__ x, unsigned short* __restrict__ xT, int H) {
    int h = blockIdx.x * 256 + threadIdx.x;
    if (h >= H) return;
    unsigned short buf[CIN];
#pragma unroll
    for (int c = 0; c < CIN; ++c) {
        __bf16 b = (__bf16)x[c * H + h];          // coalesced read across lanes
        buf[c] = __builtin_bit_cast(unsigned short, b);
    }
    uint4* dst = (uint4*)(xT + (size_t)h * CIN);  // 128B contiguous per point
    const uint4* s = (const uint4*)buf;
#pragma unroll
    for (int i = 0; i < 8; ++i) dst[i] = s[i];
}

// ------------- prolog: W (o,c,k3) f32 -> W2[o][k3*64+c] bf16 -------------
__global__ __launch_bounds__(256)
void prep_weight(const float* __restrict__ w, unsigned short* __restrict__ W2) {
    int i = blockIdx.x * 256 + threadIdx.x;
    if (i >= COUT * KTOT) return;
    int o  = i / KTOT;
    int r  = i - o * KTOT;
    int k3 = r >> 6;
    int c  = r & 63;
    __bf16 b = (__bf16)w[(o * CIN + c) * NK3 + k3];
    W2[i] = __builtin_bit_cast(unsigned short, b);
}

__global__ void zero_sums(float* __restrict__ sums) {
    if (threadIdx.x < 2 * COUT) sums[threadIdx.x] = 0.0f;
}

// ---------------- main gather-GEMM, bf16 WMMA, double-buffered ----------------
__global__ __launch_bounds__(256)
void conv_wmma(const unsigned short* __restrict__ xT,
               const unsigned short* __restrict__ W2,
               const int* __restrict__ neigh,
               float* __restrict__ out, int H)
{
    __shared__ int sIdx[NTILE * NK3];                 // 13824 B
    __shared__ unsigned int sA[2][COUT * LDSROW];     // 2 x 8704 B weight slabs
    __shared__ unsigned int sB[2][NTILE * LDSROW];    // 2 x 8704 B gathered slabs

    const int t    = threadIdx.x;
    const int lane = t & 31;
    const int wave = t >> 5;                          // 0..7
    const int n0   = blockIdx.x * NTILE;

    for (int i = t; i < NTILE * NK3; i += 256) {
        int p = i / NK3, j = i - p * NK3;
        sIdx[i] = neigh[(size_t)(n0 + p) * NK3 + j];
    }

    const v8f vzero = {0.f,0.f,0.f,0.f,0.f,0.f,0.f,0.f};
    v8f acc[2][4];                                    // [o sub-strip][n sub-tile]
#pragma unroll
    for (int a = 0; a < 2; ++a)
#pragma unroll
        for (int b = 0; b < 4; ++b) acc[a][b] = vzero;

    // wave tiling: 4 (o) x 2 (n) -> each wave = 32 o-rows x 64 n-cols
    const int obase = (wave & 3) * 32;
    const int nbase = (wave >> 2) * 64;
    const int hi    = lane >> 4;                      // half-wave id
    const int lm    = lane & 15;

    const int oLd   = t >> 1;                         // slab-load role (row 0..127)
    const int part  = t & 1;                          // 16-bf16 half of the 32-wide row

    uint4 wA0, wA1, wB0, wB1;                         // staged global data
    auto loadChunk = [&](int kc) {
        const uint4* srcA = (const uint4*)(W2 + (size_t)oLd * KTOT + kc * 32 + part * 16);
        wA0 = srcA[0]; wA1 = srcA[1];
        int idx = sIdx[oLd * NK3 + (kc >> 1)];
        const uint4* srcB = (const uint4*)(xT + (size_t)idx * CIN + (kc & 1) * 32 + part * 16);
        wB0 = srcB[0]; wB1 = srcB[1];
    };
    auto storeChunk = [&](int buf) {
        unsigned int* dA = &sA[buf][oLd * LDSROW + part * 8];
        dA[0]=wA0.x; dA[1]=wA0.y; dA[2]=wA0.z; dA[3]=wA0.w;
        dA[4]=wA1.x; dA[5]=wA1.y; dA[6]=wA1.z; dA[7]=wA1.w;
        unsigned int* dB = &sB[buf][oLd * LDSROW + part * 8];
        dB[0]=wB0.x; dB[1]=wB0.y; dB[2]=wB0.z; dB[3]=wB0.w;
        dB[4]=wB1.x; dB[5]=wB1.y; dB[6]=wB1.z; dB[7]=wB1.w;
    };

    __syncthreads();                                  // sIdx visible
    loadChunk(0);
    storeChunk(0);
    __syncthreads();

    const int k0 = hi * 4;                            // A pair offset per half-wave
    for (int kc = 0; kc < NCHUNK; ++kc) {
        const int cur = kc & 1;

        if (kc + 1 < NCHUNK) loadChunk(kc + 1);       // global loads overlap WMMAs below

        // two A fragments (16x32 bf16) for this wave's 32-row o-strip
        const unsigned int* arow0 = &sA[cur][(obase +      lm) * LDSROW];
        const unsigned int* arow1 = &sA[cur][(obase + 16 + lm) * LDSROW];
        v8u au0, au1;
        au0[0]=arow0[k0+0]; au0[1]=arow0[k0+1]; au0[2]=arow0[k0+2];  au0[3]=arow0[k0+3];
        au0[4]=arow0[k0+8]; au0[5]=arow0[k0+9]; au0[6]=arow0[k0+10]; au0[7]=arow0[k0+11];
        au1[0]=arow1[k0+0]; au1[1]=arow1[k0+1]; au1[2]=arow1[k0+2];  au1[3]=arow1[k0+3];
        au1[4]=arow1[k0+8]; au1[5]=arow1[k0+9]; au1[6]=arow1[k0+10]; au1[7]=arow1[k0+11];
        v16bf af0 = __builtin_bit_cast(v16bf, au0);
        v16bf af1 = __builtin_bit_cast(v16bf, au1);

#pragma unroll
        for (int b = 0; b < 4; ++b) {
            const unsigned int* brow = &sB[cur][(nbase + b * 16 + lm) * LDSROW + hi * 4];
            v8u bu;
            bu[0]=brow[0]; bu[1]=brow[1]; bu[2]=brow[2]; bu[3]=brow[3];
            bu[4]=brow[8]; bu[5]=brow[9]; bu[6]=brow[10]; bu[7]=brow[11];
            v16bf bf = __builtin_bit_cast(v16bf, bu);
            acc[0][b] = __builtin_amdgcn_wmma_f32_16x16x32_bf16(
                false, af0, false, bf, (short)0, acc[0][b], false, false);
            acc[1][b] = __builtin_amdgcn_wmma_f32_16x16x32_bf16(
                false, af1, false, bf, (short)0, acc[1][b], false, false);
        }

        if (kc + 1 < NCHUNK) storeChunk(1 - cur);     // fill the other buffer
        __syncthreads();                              // one barrier per chunk
    }

    // write raw conv output: D element (m = v + 8*hi, n = lm) per C/D VGPR layout
#pragma unroll
    for (int a = 0; a < 2; ++a) {
#pragma unroll
        for (int b = 0; b < 4; ++b) {
            int n = n0 + nbase + b * 16 + lm;
#pragma unroll
            for (int v = 0; v < 8; ++v) {
                int m = obase + a * 16 + hi * 8 + v;
                out[(size_t)m * H + n] = acc[a][b][v];
            }
        }
    }
}

// ---------------- per-channel sum / sum-of-squares ----------------
__global__ __launch_bounds__(256)
void reduce_stats(const float* __restrict__ out, float* __restrict__ sums, int H) {
    const int o = blockIdx.x;
    const int per = H / gridDim.y;
    const float* p = out + (size_t)o * H + (size_t)blockIdx.y * per;
    float s = 0.f, s2 = 0.f;
    for (int i = threadIdx.x; i < per; i += 256) {
        float v = p[i];
        s += v; s2 += v * v;
    }
    __shared__ float rs[256], rq[256];
    rs[threadIdx.x] = s; rq[threadIdx.x] = s2;
    __syncthreads();
    for (int off = 128; off > 0; off >>= 1) {
        if (threadIdx.x < off) {
            rs[threadIdx.x] += rs[threadIdx.x + off];
            rq[threadIdx.x] += rq[threadIdx.x + off];
        }
        __syncthreads();
    }
    if (threadIdx.x == 0) {
        atomicAdd(&sums[o], rs[0]);
        atomicAdd(&sums[COUT + o], rq[0]);
    }
}

// ---------------- BN + ReLU epilogue (in place, float4) ----------------
__global__ __launch_bounds__(256)
void bn_relu(float* __restrict__ out, const float* __restrict__ sums,
             const float* __restrict__ gamma, const float* __restrict__ beta, int H) {
    size_t i = (size_t)blockIdx.x * 256 + threadIdx.x;
    size_t total4 = (size_t)COUT * H / 4;
    if (i >= total4) return;
    int o = (int)((i * 4) / H);                      // H % 4 == 0, no straddle
    float invH  = 1.0f / (float)H;
    float mean  = sums[o] * invH;
    float var   = sums[COUT + o] * invH - mean * mean;
    float scale = rsqrtf(var + 1e-5f) * gamma[o];
    float bias  = beta[o] - mean * scale;
    float4* p = (float4*)out;
    float4 v = p[i];
    v.x = fmaxf(v.x * scale + bias, 0.f);
    v.y = fmaxf(v.y * scale + bias, 0.f);
    v.z = fmaxf(v.z * scale + bias, 0.f);
    v.w = fmaxf(v.w * scale + bias, 0.f);
    p[i] = v;
}

extern "C" void kernel_launch(void* const* d_in, const int* in_sizes, int n_in,
                              void* d_out, int out_size, void* d_ws, size_t ws_size,
                              hipStream_t stream) {
    const float* x      = (const float*)d_in[0];   // (1,64,H,1)
    const int*   neigh  = (const int*)  d_in[1];   // (H,27)
    const float* weight = (const float*)d_in[2];   // (128,64,27)
    const float* gamma  = (const float*)d_in[3];
    const float* beta   = (const float*)d_in[4];
    float* out = (float*)d_out;

    const int H = in_sizes[0] / CIN;               // 65536

    char* ws = (char*)d_ws;
    unsigned short* xT = (unsigned short*)ws;                       // H*64 bf16 = 8 MB
    size_t xtBytes = (size_t)H * CIN * 2;
    unsigned short* W2 = (unsigned short*)(ws + xtBytes);           // 128*1728 bf16
    size_t w2Bytes = (size_t)COUT * KTOT * 2;
    float* sums = (float*)(ws + xtBytes + w2Bytes);                 // 256 f32

    prep_x<<<(H + 255) / 256, 256, 0, stream>>>(x, xT, H);
    prep_weight<<<(COUT * KTOT + 255) / 256, 256, 0, stream>>>(weight, W2);
    zero_sums<<<1, 256, 0, stream>>>(sums);

    conv_wmma<<<H / NTILE, 256, 0, stream>>>(xT, W2, neigh, out, H);

    dim3 rgrid(COUT, 64);
    reduce_stats<<<rgrid, 256, 0, stream>>>(out, sums, H);

    size_t total4 = (size_t)COUT * H / 4;
    bn_relu<<<(unsigned)((total4 + 255) / 256), 256, 0, stream>>>(out, sums, gamma, beta, H);
}
